// TiledAttentionModule_15668040695927
// MI455X (gfx1250) — compile-verified, matrix operand order checked by reading, not run on hardware
//
#include <hip/hip_runtime.h>

// ---------------------------------------------------------------------------
// Causal MHA block for MI455X (gfx1250, wave32, WMMA f32_16x16x32_f16)
//   B=2, T=2048, D_MODEL=2048, H=16, D_HEAD=128
// Pipeline:
//   1) Q = (x @ Wq^T) * scale, K = x @ Wk^T          (f32 in -> f16, WMMA)
//      Vt = transpose_per_head(x @ Wv^T)             (f16 [bh, d, t] layout)
//   2) flash attention per (b,h), 128-row query blocks, online softmax,
//      double-buffered async K/Vt tile staging (ASYNCcnt)
//   3) out = O @ Wo^T                                (f16 in -> f32 out, WMMA)
// ---------------------------------------------------------------------------

typedef _Float16 half_t;
typedef __attribute__((ext_vector_type(16))) _Float16 v16h;
typedef __attribute__((ext_vector_type(8)))  _Float16 v8h;
typedef __attribute__((ext_vector_type(8)))  float    v8f;
typedef __attribute__((ext_vector_type(4)))  float    v4f;
typedef int v4i __attribute__((vector_size(16)));  // matches builtin param type

#define D_MODEL 2048
#define SEQ_T   2048
#define N_HEADS 16
#define D_HEAD  128
#define ATT_SCALE 0.08838834764831845f  // 1/sqrt(128)

// ---- gfx1250 async global->LDS copy (guarded; sync fallback) ---------------
#if __has_builtin(__builtin_amdgcn_global_load_async_to_lds_b128) && \
    __has_builtin(__builtin_amdgcn_s_wait_asynccnt)
#define HAVE_ASYNC_LDS 1
#else
#define HAVE_ASYNC_LDS 0
#endif

__device__ __forceinline__ void async_copy_b128(half_t* lds_dst,
                                                const half_t* gsrc) {
#if HAVE_ASYNC_LDS
  v4i* g = (v4i*)gsrc;   // drop const + reinterpret (generic)
  v4i* l = (v4i*)lds_dst;
  __builtin_amdgcn_global_load_async_to_lds_b128(
      (__attribute__((address_space(1))) v4i*)g,
      (__attribute__((address_space(3))) v4i*)l, 0, 0);
#else
  *(v8h*)lds_dst = *(const v8h*)gsrc;
#endif
}

__device__ __forceinline__ void async_wait_le16() {
#if HAVE_ASYNC_LDS
  __builtin_amdgcn_s_wait_asynccnt(16);
#endif
}
__device__ __forceinline__ void async_wait_0() {
#if HAVE_ASYNC_LDS
  __builtin_amdgcn_s_wait_asynccnt(0);
#endif
}

// ---- WMMA fragment load from LDS -------------------------------------------
// A-fragment (16x32, f16): lane l<16 -> row (l&15), K {k0..k0+7, k0+16..k0+23}
//                          lane l>=16 -> row (l&15), K {k0+8..15, k0+24..31}
// B-fragment (32x16) uses identical per-lane addressing with "row" = output
// column (B is always arranged so its columns are contiguous rows in LDS).
__device__ __forceinline__ v16h ld_frag(const half_t* lds, int stride,
                                        int rbase, int kbase) {
  const int l = threadIdx.x & 31;
  const half_t* p = lds + (rbase + (l & 15)) * stride + kbase + ((l >> 4) << 3);
  v8h lo = *(const v8h*)(p);
  v8h hi = *(const v8h*)(p + 16);
  return __builtin_shufflevector(lo, hi, 0, 1, 2, 3, 4, 5, 6, 7,
                                 8, 9, 10, 11, 12, 13, 14, 15);
}

__device__ __forceinline__ v8f wmma_f16(v16h a, v16h b, v8f c) {
  return __builtin_amdgcn_wmma_f32_16x16x32_f16(false, a, false, b,
                                                (short)0, c, false, false);
}

// convert 16 consecutive f32 -> 16 f16 in LDS (two ds_store_b128)
__device__ __forceinline__ void cvt_store16(half_t* dst, const float* src) {
  v4f a = *(const v4f*)(src);
  v4f b = *(const v4f*)(src + 4);
  v4f c = *(const v4f*)(src + 8);
  v4f d = *(const v4f*)(src + 12);
  v8h h0, h1;
#pragma unroll
  for (int i = 0; i < 4; ++i) {
    h0[i]     = (_Float16)a[i];
    h0[i + 4] = (_Float16)b[i];
    h1[i]     = (_Float16)c[i];
    h1[i + 4] = (_Float16)d[i];
  }
  *(v8h*)(dst)     = h0;
  *(v8h*)(dst + 8) = h1;
}

// ---------------------------------------------------------------------------
// GEMM:  C[M,N] = (A[M,K] @ W[N,K]^T) * out_scale
//   A_HALF : A is f16 (else f32, converted while staging)
//   OUT_HALF: C written as f16 (else f32)
//   OUT_VT : write per-head transposed f16 layout Vt[(b*16+h)*128+d][t]
// Block: 256 threads = 8 waves; block tile 128x128; wave tile 32x64.
// ---------------------------------------------------------------------------
#define KTILE 64
#define KSTR  72  // 64 + 8 halves pad; row stride = 144 B (16-B aligned)

template <bool A_HALF, bool OUT_HALF, bool OUT_VT>
__global__ __launch_bounds__(256) void gemm_xWT(const void* __restrict__ Ap,
                                                const float* __restrict__ W,
                                                void* __restrict__ Cp,
                                                int M, int N, int K,
                                                float out_scale) {
  __shared__ alignas(16) half_t sA[128 * KSTR];
  __shared__ alignas(16) half_t sB[128 * KSTR];

  const int tid  = threadIdx.x;
  const int lane = tid & 31;
  const int wid  = tid >> 5;
  const int wave_m = (wid & 3) * 32;  // 4 waves along M
  const int wave_n = (wid >> 2) * 64; // 2 waves along N
  const int m0 = blockIdx.y * 128;
  const int n0 = blockIdx.x * 128;

  const int rowH   = tid >> 2;        // 0..63
  const int colSeg = (tid & 3) * 16;  // 0,16,32,48

  v8f acc[2][4] = {};

  for (int k0 = 0; k0 < K; k0 += KTILE) {
#pragma unroll
    for (int pass = 0; pass < 2; ++pass) {
      const int row = rowH + pass * 64;
      // stage A tile
      if constexpr (A_HALF) {
        const half_t* a = (const half_t*)Ap + (size_t)(m0 + row) * K + k0 + colSeg;
        async_copy_b128(&sA[row * KSTR + colSeg], a);
        async_copy_b128(&sA[row * KSTR + colSeg + 8], a + 8);
        if (k0 + KTILE < K) __builtin_prefetch(a + KTILE, 0, 1);
      } else {
        const float* a = (const float*)Ap + (size_t)(m0 + row) * K + k0 + colSeg;
        cvt_store16(&sA[row * KSTR + colSeg], a);
        if (k0 + KTILE < K) __builtin_prefetch(a + KTILE, 0, 1);
      }
      // stage B tile: rows of W (= columns of W^T), contiguous in K
      const float* w = W + (size_t)(n0 + row) * K + k0 + colSeg;
      cvt_store16(&sB[row * KSTR + colSeg], w);
      if (k0 + KTILE < K) __builtin_prefetch(w + KTILE, 0, 1);
    }
    if constexpr (A_HALF) async_wait_0();
    __syncthreads();

#pragma unroll
    for (int kk = 0; kk < KTILE; kk += 32) {
      v16h aF[2], bF[4];
      aF[0] = ld_frag(sA, KSTR, wave_m, kk);
      aF[1] = ld_frag(sA, KSTR, wave_m + 16, kk);
#pragma unroll
      for (int j = 0; j < 4; ++j) bF[j] = ld_frag(sB, KSTR, wave_n + j * 16, kk);
#pragma unroll
      for (int i = 0; i < 2; ++i)
#pragma unroll
        for (int j = 0; j < 4; ++j) acc[i][j] = wmma_f16(aF[i], bF[j], acc[i][j]);
    }
    __syncthreads();
  }

  // epilogue: C/D layout -> row = r + 8*(lane>=16), col = lane&15
  const int colc   = lane & 15;
  const int rowAdd = (lane >> 4) << 3;
#pragma unroll
  for (int i = 0; i < 2; ++i)
#pragma unroll
    for (int j = 0; j < 4; ++j) {
      const int rbase = m0 + wave_m + i * 16 + rowAdd;
      const int col   = n0 + wave_n + j * 16 + colc;
#pragma unroll
      for (int r = 0; r < 8; ++r) {
        const float v = acc[i][j][r] * out_scale;
        if constexpr (OUT_VT) {
          // token row -> (b, t); model col -> (h, d); write Vt[(b*16+h)*128+d][t]
          const int row = rbase + r;
          const int bb  = row >> 11;         // / SEQ_T
          const int t   = row & (SEQ_T - 1);
          const int hh  = col >> 7;          // / D_HEAD
          const int dd  = col & (D_HEAD - 1);
          ((half_t*)Cp)[(((size_t)(bb * N_HEADS + hh) * D_HEAD + dd) * SEQ_T) + t] =
              (_Float16)v;
        } else if constexpr (OUT_HALF) {
          ((half_t*)Cp)[(size_t)(rbase + r) * N + col] = (_Float16)v;
        } else {
          ((float*)Cp)[(size_t)(rbase + r) * N + col] = v;
        }
      }
    }
}

// ---------------------------------------------------------------------------
// Flash attention, causal, one (b,h) x 128-query-row block per workgroup.
// 8 waves; wave w owns query rows [w*16, w*16+16). Online softmax operates
// directly on the WMMA C-layout (a row lives across one 16-lane half).
// K and Vt tiles are double-buffered and streamed with async global->LDS
// copies (16 b128 ops per thread per tile; ASYNCcnt completes in order).
// ---------------------------------------------------------------------------
#define ASTR 136  // 128 + 8 halves pad; row stride 272 B (16-B aligned)

__device__ __forceinline__ void issue_kv_tile(const half_t* __restrict__ Kh,
                                              const half_t* __restrict__ Vt,
                                              half_t* sKbuf, half_t* sVbuf,
                                              int b, int bh, int col0, int j,
                                              int tid) {
  const int row = tid >> 1;            // K: key row / Vt: d row
  const int cs  = (tid & 1) * 64;
  const half_t* ks = Kh + ((size_t)b * SEQ_T + (size_t)j * 128 + row) * D_MODEL +
                     col0 + cs;
  half_t* kd = sKbuf + row * ASTR + cs;
  const half_t* vs = Vt + ((size_t)bh * D_HEAD + row) * SEQ_T + (size_t)j * 128 + cs;
  half_t* vd = sVbuf + row * ASTR + cs;
#pragma unroll
  for (int c = 0; c < 64; c += 8) {
    async_copy_b128(kd + c, ks + c);
    async_copy_b128(vd + c, vs + c);
  }
}

__global__ __launch_bounds__(256) void flash_attn(const half_t* __restrict__ Qh,
                                                  const half_t* __restrict__ Kh,
                                                  const half_t* __restrict__ Vt,
                                                  half_t* __restrict__ Oh) {
  __shared__ alignas(16) half_t sQ[128 * ASTR];
  __shared__ alignas(16) half_t sK[2][128 * ASTR];
  __shared__ alignas(16) half_t sVt[2][128 * ASTR];
  __shared__ alignas(16) half_t sP[128 * ASTR];

  const int tid  = threadIdx.x;
  const int lane = tid & 31;
  const int wid  = tid >> 5;
  const int qi = blockIdx.x;          // query block 0..15
  const int bh = blockIdx.y;          // 0..31
  const int b  = bh >> 4;
  const int h  = bh & 15;
  const size_t rowQ0 = (size_t)b * SEQ_T + (size_t)qi * 128;
  const int col0 = h * D_HEAD;

  // ---- issue Q (already pre-scaled by the Q-GEMM) + first K/Vt tile ----
  {
    const int row = tid >> 1;
    const int cs  = (tid & 1) * 64;
    const half_t* src = Qh + (rowQ0 + row) * D_MODEL + col0 + cs;
    half_t* dst = &sQ[row * ASTR + cs];
#pragma unroll
    for (int c = 0; c < 64; c += 8) async_copy_b128(dst + c, src + c);
  }
  issue_kv_tile(Kh, Vt, sK[0], sVt[0], b, bh, col0, 0, tid);

  v8f o_acc[8] = {};
  float mstate[8], lstate[8];
#pragma unroll
  for (int r = 0; r < 8; ++r) { mstate[r] = -1e30f; lstate[r] = 0.f; }

  for (int j = 0; j <= qi; ++j) {
    const int bi = j & 1;
    __syncthreads();  // all waves finished reading buffer bi^1 (tile j-1)
    if (j < qi) {
      issue_kv_tile(Kh, Vt, sK[bi ^ 1], sVt[bi ^ 1], b, bh, col0, j + 1, tid);
      async_wait_le16();  // newest 16 ops may be in flight; tile j (and Q) done
    } else {
      async_wait_0();
    }
    __syncthreads();  // every wave's tile-j portion is visible in LDS

    // ---- S = Q K^T  (wave strip: 16 q-rows x 128 keys, 8 accum tiles) ----
    v8f s_acc[8] = {};
#pragma unroll
    for (int kk = 0; kk < D_HEAD; kk += 32) {
      v16h aQ = ld_frag(sQ, ASTR, wid * 16, kk);
#pragma unroll
      for (int t = 0; t < 8; ++t) {
        v16h bK = ld_frag(sK[bi], ASTR, t * 16, kk);
        s_acc[t] = wmma_f16(aQ, bK, s_acc[t]);
      }
    }

    // ---- causal mask on the diagonal block ----
    if (j == qi) {
      const int qrow = wid * 16 + ((lane >> 4) << 3);
      const int kcol = lane & 15;
#pragma unroll
      for (int t = 0; t < 8; ++t) {
        const int kg = t * 16 + kcol;
#pragma unroll
        for (int r = 0; r < 8; ++r)
          if (kg > qrow + r) s_acc[t][r] = -1e30f;
      }
    }

    // ---- online softmax: a row lives across one 16-lane half ----
    float mnew[8], scl[8];
#pragma unroll
    for (int r = 0; r < 8; ++r) {
      float mx = s_acc[0][r];
#pragma unroll
      for (int t = 1; t < 8; ++t) mx = fmaxf(mx, s_acc[t][r]);
#pragma unroll
      for (int d = 1; d < 16; d <<= 1) mx = fmaxf(mx, __shfl_xor(mx, d, 32));
      mnew[r] = fmaxf(mstate[r], mx);
      scl[r]  = __expf(mstate[r] - mnew[r]);
      mstate[r] = mnew[r];
    }
#pragma unroll
    for (int r = 0; r < 8; ++r) {
      float psum = 0.f;
#pragma unroll
      for (int t = 0; t < 8; ++t) {
        float p = __expf(s_acc[t][r] - mnew[r]);
        s_acc[t][r] = p;
        psum += p;
      }
#pragma unroll
      for (int d = 1; d < 16; d <<= 1) psum += __shfl_xor(psum, d, 32);
      lstate[r] = lstate[r] * scl[r] + psum;
    }

    // ---- P -> LDS (own 16-row strip; DS in-order within wave),
    //      rescale O accumulators ----
    {
      const int rowb = wid * 16 + ((lane >> 4) << 3);
      const int colc = lane & 15;
#pragma unroll
      for (int t = 0; t < 8; ++t)
#pragma unroll
        for (int r = 0; r < 8; ++r) {
          sP[(rowb + r) * ASTR + t * 16 + colc] = (_Float16)s_acc[t][r];
          o_acc[t][r] *= scl[r];
        }
    }

    // ---- O += P @ V  (B columns are contiguous rows of sVt) ----
#pragma unroll
    for (int kk = 0; kk < 128; kk += 32) {
      v16h aP = ld_frag(sP, ASTR, wid * 16, kk);
#pragma unroll
      for (int t = 0; t < 8; ++t) {
        v16h bV = ld_frag(sVt[bi], ASTR, t * 16, kk);
        o_acc[t] = wmma_f16(aP, bV, o_acc[t]);
      }
    }
  }

  // ---- normalize and write O (f16, natural [B,T,H,Dh] == [B,T,D] layout) ----
  {
    const int rowb = wid * 16 + ((lane >> 4) << 3);
    const int colc = lane & 15;
#pragma unroll
    for (int t = 0; t < 8; ++t)
#pragma unroll
      for (int r = 0; r < 8; ++r) {
        const float v = o_acc[t][r] / lstate[r];
        Oh[(rowQ0 + rowb + r) * D_MODEL + col0 + t * 16 + colc] = (_Float16)v;
      }
  }
}

// ---------------------------------------------------------------------------
extern "C" void kernel_launch(void* const* d_in, const int* in_sizes, int n_in,
                              void* d_out, int out_size, void* d_ws, size_t ws_size,
                              hipStream_t stream) {
  const float* x  = (const float*)d_in[0];
  const float* Wq = (const float*)d_in[1];
  const float* Wk = (const float*)d_in[2];
  const float* Wv = (const float*)d_in[3];
  const float* Wo = (const float*)d_in[4];
  float* out = (float*)d_out;

  const int M = 2 * SEQ_T;      // 4096 token rows
  const int N = D_MODEL;        // 2048
  const int K = D_MODEL;        // 2048
  const size_t elems = (size_t)M * D_MODEL;

  half_t* Qh = (half_t*)d_ws;
  half_t* Kh = Qh + elems;
  half_t* Vt = Kh + elems;      // per-head transposed: [(b*16+h)*128 + d][t]
  half_t* Oh = Vt + elems;

  dim3 blk(256);
  dim3 gproj(N / 128, M / 128);   // 16 x 32

  gemm_xWT<false, true, false><<<gproj, blk, 0, stream>>>(x, Wq, Qh, M, N, K,
                                                          ATT_SCALE);
  gemm_xWT<false, true, false><<<gproj, blk, 0, stream>>>(x, Wk, Kh, M, N, K, 1.f);
  gemm_xWT<false, true, true><<<gproj, blk, 0, stream>>>(x, Wv, Vt, M, N, K, 1.f);

  dim3 gattn(SEQ_T / 128, 2 * N_HEADS);  // 16 x 32
  flash_attn<<<gattn, blk, 0, stream>>>(Qh, Kh, Vt, Oh);

  gemm_xWT<true, false, false><<<gproj, blk, 0, stream>>>(Oh, Wo, out, M, N, K, 1.f);
}
